// EwaldBlock_57904749085225
// MI455X (gfx1250) — matrix operand order repaired and checked by reading, not compiled, not run on hardware
//
#include <hip/hip_runtime.h>
#include <hip/hip_bf16.h>

// ---------------------------------------------------------------------------
// Types for CDNA5 WMMA (gfx1250, wave32)
// ---------------------------------------------------------------------------
typedef __attribute__((ext_vector_type(16))) __bf16        v16bf;
typedef __attribute__((ext_vector_type(8)))  float         v8f;
typedef __attribute__((ext_vector_type(4)))  unsigned int  u32x4;
typedef __attribute__((ext_vector_type(4)))  float         f32x4;

union FragU {
    v16bf          v;
    u32x4          q[2];
    unsigned int   w[8];
    unsigned short u[16];
};

// RNE convert (prep kernels only)
static __device__ __forceinline__ unsigned short f2bf(float f) {
    unsigned int u = __float_as_uint(f);
    u += 0x7fffu + ((u >> 16) & 1u);
    return (unsigned short)(u >> 16);
}
// Truncating convert: single shift, folds into *_store_d16_hi_b16
static __device__ __forceinline__ unsigned short bft(float f) {
    return (unsigned short)(__float_as_uint(f) >> 16);
}
// Pack two floats -> two bf16 in one v_perm_b32
static __device__ __forceinline__ unsigned int pk2bf(float lo, float hi) {
    return __builtin_amdgcn_perm(__float_as_uint(hi), __float_as_uint(lo),
                                 0x07060302u);
}
static __device__ __forceinline__ float bf2f(unsigned short h) {
    return __uint_as_float(((unsigned int)h) << 16);
}
static __device__ __forceinline__ float silu_f(float x) {
    return x * (1.0f / (1.0f + __expf(-x)));
}

// Fragment-order offset for a KxN bf16 matrix consumed as WMMA B:
// block (k>>5, n>>4) of 512 elems; inside: lane=(n&15)+(k&16), idx=k&15.
static __device__ __forceinline__ size_t frag_off(int k, int n, int ncols) {
    return (size_t)(((k >> 5) * (ncols >> 4) + (n >> 4)) << 9)
         + (size_t)((((n & 15) + (k & 16)) << 4) + (k & 15));
}

// A fragment (16x32 bf16): lane l<16 -> row l, K={0..15};
// lane l>=16 -> row l-16, K={16..31}. LDS row stride 40 ushorts (80 B)
// keeps both halves 16B-aligned -> 2x ds_load_b128, conflict-free.
static __device__ __forceinline__ v16bf load_a_frag40(const unsigned short* smem,
                                                      int rowBase) {
    int lane = threadIdx.x & 31;
    const unsigned short* p = smem + (rowBase + (lane & 15)) * 40
                                   + ((lane < 16) ? 0 : 16);
    FragU f;
    f.q[0] = *(const u32x4*)(p);
    f.q[1] = *(const u32x4*)(p + 8);
    return f.v;
}

// 32 contiguous aligned bytes per lane (LDS or global fragment-order tile)
static __device__ __forceinline__ v16bf load_frag32B(const unsigned short* p) {
    FragU f;
    f.q[0] = *(const u32x4*)(p);
    f.q[1] = *(const u32x4*)(p + 8);
    return f.v;
}

static __device__ __forceinline__ v8f wmma_bf16(v16bf a, v16bf b, v8f c) {
    return __builtin_amdgcn_wmma_f32_16x16x32_bf16(false, a, false, b,
                                                   (short)0, c, false, false);
}

// Async global->LDS 16B copy (GLOBAL_LOAD_ASYNC_TO_LDS_B128, ASYNCcnt).
// VDST = LDS byte address, VADDR = 64-bit global address (GV mode).
static __device__ __forceinline__ void async_copy_b128(const unsigned short* gsrc,
                                                       unsigned short* ldst) {
    unsigned int lds = (unsigned int)(size_t)ldst;
    unsigned long long ga = (unsigned long long)(size_t)gsrc;
    asm volatile("global_load_async_to_lds_b128 %0, %1, off"
                 :: "v"(lds), "v"(ga) : "memory");
}
static __device__ __forceinline__ void wait_asynccnt0() {
    asm volatile("s_wait_asynccnt 0" ::: "memory");
}

// Scatter 8 bf16 (consecutive n, fixed k) into fragment-order LDS tile.
static __device__ __forceinline__ void scatter_bfrag(unsigned short* smem,
                                                     int kk, int eb,
                                                     const unsigned short* vals) {
    unsigned short* dst = smem + ((((eb >> 4) << 5) + (eb & 15) + (kk & 16)) << 4)
                               + (kk & 15);
#pragma unroll
    for (int i = 0; i < 8; ++i) dst[i << 4] = vals[i];
}

// ---------------------------------------------------------------------------
// Prep kernels
// ---------------------------------------------------------------------------
__global__ void seg_offsets_kernel(const int* __restrict__ bseg,
                                   int* __restrict__ seg, int N, int B) {
    int n = blockIdx.x * blockDim.x + threadIdx.x;
    if (n == 0) {
        int b0 = bseg[0];
        for (int b = 0; b <= b0; ++b) seg[b] = 0;
        int bl = bseg[N - 1];
        for (int b = bl + 1; b <= B; ++b) seg[b] = N;
    }
    if (n >= 1 && n < N) {
        int p = bseg[n - 1], q = bseg[n];
        for (int b = p + 1; b <= q; ++b) seg[b] = n;
    }
}

// Bulk fp32 -> bf16 row-major convert (8 elements / thread, v_perm packing)
__global__ void f32_to_bf16_kernel(const float* __restrict__ src,
                                   unsigned short* __restrict__ dst, int n8) {
    int i = blockIdx.x * blockDim.x + threadIdx.x;
    if (i >= n8) return;
    const f32x4* s = (const f32x4*)(src + (size_t)i * 8);
    f32x4 a = s[0], b = s[1];
    u32x4 o;
    o.x = pk2bf(a.x, a.y);
    o.y = pk2bf(a.z, a.w);
    o.z = pk2bf(b.x, b.y);
    o.w = pk2bf(b.z, b.w);
    *(u32x4*)(dst + (size_t)i * 8) = o;
}

// Weight -> bf16, transposed AND laid out in WMMA-B fragment order so GEMMs
// load B fragments directly from global with coalesced b128 loads.
__global__ void wtrans_kernel(const float* __restrict__ W,
                              unsigned short* __restrict__ Wt, int E) {
    int idx = blockIdx.x * blockDim.x + threadIdx.x;
    if (idx >= E * E) return;
    int e = idx / E, j = idx % E;          // j = reduction dim, e = output col
    Wt[frag_off(j, e, E)] = f2bf(W[idx]);
}

// kf[k,e] = 0.01 * sum_d W_up[e,d] * W_down[d,k]   (0.01 folded in here)
__global__ void kfilter_kernel(const float* __restrict__ Wup,
                               const float* __restrict__ Wdown,
                               float* __restrict__ kf, int K, int E, int D) {
    int idx = blockIdx.x * blockDim.x + threadIdx.x;
    if (idx >= K * E) return;
    int kk = idx / E, e = idx % E;
    float a = 0.0f;
#pragma unroll
    for (int d = 0; d < 8; ++d) a += Wup[e * D + d] * Wdown[d * K + kk];
    kf[idx] = 0.01f * a;
}

// dot[n,k] (fp32 output #2) + bf16 cos/sin in row- and k-major layouts
__global__ void dot_cs_kernel(const float* __restrict__ x,
                              const float* __restrict__ kvec,
                              const int* __restrict__ bseg,
                              float* __restrict__ dot_out,
                              unsigned short* __restrict__ cM,
                              unsigned short* __restrict__ sM,
                              unsigned short* __restrict__ cT,
                              unsigned short* __restrict__ sT,
                              int N, int K) {
    int idx = blockIdx.x * blockDim.x + threadIdx.x;
    if (idx >= N * K) return;
    int n = idx / K, kk = idx % K;
    int b = bseg[n];
    const float* kp = kvec + ((size_t)b * K + kk) * 3;
    float d = kp[0] * x[n * 3 + 0] + kp[1] * x[n * 3 + 1] + kp[2] * x[n * 3 + 2];
    dot_out[idx] = d;
    unsigned short cb = bft(cosf(d)), sb = bft(sinf(d));
    cM[(size_t)n * K + kk] = cb;
    sM[(size_t)n * K + kk] = sb;
    cT[(size_t)kk * N + n] = cb;
    sT[(size_t)kk * N + n] = sb;
}

// ---------------------------------------------------------------------------
// bf16 WMMA GEMM with double-buffered ASYNC A-tile (global -> LDS, no VGPR
// staging) and fragment-order B weights read straight from global.
// out[m,n] = [skip[m,n] +] silu( sum_k A[m,k] * W^T[k,n] )
// Block = 256 threads (8 waves), tile 128x64, K-step 32, 1 barrier / step.
// ---------------------------------------------------------------------------
__global__ __launch_bounds__(256) void gemm_silu_kernel(
    const unsigned short* __restrict__ Ab,
    const unsigned short* __restrict__ BtF,
    const float* __restrict__ skipF, const unsigned short* __restrict__ skipB,
    float* __restrict__ outF, unsigned short* __restrict__ outB,
    int M, int Ncols, int Kd) {
    __shared__ __align__(16) unsigned short As[2][128 * 40];
    int m0 = blockIdx.y * 128, n0 = blockIdx.x * 64;
    int t = threadIdx.x, w = t >> 5, lane = t & 31;
    int ntiles = Ncols >> 4;
    v8f acc[4] = {};

    int rr = t >> 1, jb = (t & 1) * 16;
    const unsigned short* gA = Ab + (size_t)(m0 + rr) * Kd + jb;   // += kt later
    unsigned short* lA[2] = { &As[0][rr * 40 + jb], &As[1][rr * 40 + jb] };

    // Prologue: async-copy tile 0 into buffer 0 (32 B per thread).
    async_copy_b128(gA,     lA[0]);
    async_copy_b128(gA + 8, lA[0] + 8);

    int buf = 0;
    for (int kt = 0; kt < Kd; kt += 32) {
        wait_asynccnt0();       // my copies into As[buf] are done
        __syncthreads();        // everyone's are done; prior reads retired
        if (kt + 32 < Kd) {     // async-prefetch next tile into other buffer
            const unsigned short* gn = gA + kt + 32;
            async_copy_b128(gn,     lA[buf ^ 1]);
            async_copy_b128(gn + 8, lA[buf ^ 1] + 8);
        }
        // B fragments straight from global: wave reads 4 contiguous-1KB tiles
        const unsigned short* bp = BtF
            + ((size_t)((kt >> 5) * ntiles + (n0 >> 4)) << 9) + (lane << 4);
        v16bf b0 = load_frag32B(bp + (0 << 9));
        v16bf b1 = load_frag32B(bp + (1 << 9));
        v16bf b2 = load_frag32B(bp + (2 << 9));
        v16bf b3 = load_frag32B(bp + (3 << 9));
        v16bf a  = load_a_frag40(&As[buf][0], w * 16);
        acc[0] = wmma_bf16(a, b0, acc[0]);
        acc[1] = wmma_bf16(a, b1, acc[1]);
        acc[2] = wmma_bf16(a, b2, acc[2]);
        acc[3] = wmma_bf16(a, b3, acc[3]);
        buf ^= 1;
    }

#pragma unroll
    for (int nt = 0; nt < 4; ++nt)
#pragma unroll
        for (int r = 0; r < 8; ++r) {
            int row = m0 + w * 16 + ((lane < 16) ? r : r + 8);
            int col = n0 + nt * 16 + (lane & 15);
            size_t o = (size_t)row * Ncols + col;
            float v = silu_f(acc[nt][r]);
            if (skipF)      v += skipF[o];
            else if (skipB) v += bf2f(skipB[o]);
            if (outF) outF[o] = v;
            else      outB[o] = bft(v);
        }
}

// ---------------------------------------------------------------------------
// Structure factors, per structure b:
//   Fr[b,k,e] = kf[k,e] * sum_{n in seg b} c[n,k]*hres[n,e]   (Fi with sin)
// Atom loop starts 32-aligned; both segment ends zero-masked in registers.
// Fr/Fi are written in fragment order for direct consumption by hupdate.
// grid = (E/64, B).
// ---------------------------------------------------------------------------
__global__ __launch_bounds__(256) void sf_kernel(
    const unsigned short* __restrict__ cT, const unsigned short* __restrict__ sT,
    const unsigned short* __restrict__ hres, const float* __restrict__ kf,
    const int* __restrict__ seg,
    unsigned short* __restrict__ Fr, unsigned short* __restrict__ Fi,
    int N, int K, int E) {
    __shared__ __align__(16) unsigned short Cs[128 * 40];
    __shared__ __align__(16) unsigned short Ss[128 * 40];
    __shared__ __align__(16) unsigned short Hs[4 * 32 * 16];
    int b  = blockIdx.y;
    int e0 = blockIdx.x * 64;
    int s0 = seg[b], s1 = seg[b + 1];
    int t = threadIdx.x, w = t >> 5, lane = t & 31;
    v8f accR[4] = {};
    v8f accI[4] = {};

    for (int a0 = (s0 & ~31); a0 < s1; a0 += 32) {
        __syncthreads();
        {   // A tiles (cT,sT): 128(k) x 32(atoms), aligned vec loads + masking
            int kr = t >> 1, jb = (t & 1) * 16;
            int base = a0 + jb;
            const u32x4* pc = (const u32x4*)(cT + (size_t)kr * N + base);
            const u32x4* ps = (const u32x4*)(sT + (size_t)kr * N + base);
            FragU fc, fs;
            fc.q[0] = pc[0]; fc.q[1] = pc[1];
            fs.q[0] = ps[0]; fs.q[1] = ps[1];
#pragma unroll
            for (int i = 0; i < 16; ++i) {
                int n = base + i;
                if (n < s0 || n >= s1) { fc.u[i] = 0; fs.u[i] = 0; }
            }
            u32x4* dc = (u32x4*)(Cs + kr * 40 + jb);
            u32x4* ds = (u32x4*)(Ss + kr * 40 + jb);
            dc[0] = fc.q[0]; dc[1] = fc.q[1];
            ds[0] = fs.q[0]; ds[1] = fs.q[1];
        }
        {   // B tile (hres): 32(atoms) x 64(e), fragment-order scatter
            int j = t >> 3, eb = (t & 7) * 8;
            int n = a0 + j;
            unsigned short vals[8] = {0, 0, 0, 0, 0, 0, 0, 0};
            if (n >= s0 && n < s1) {
                u32x4 raw = *(const u32x4*)(hres + (size_t)n * E + e0 + eb);
                const unsigned short* rp = (const unsigned short*)&raw;
#pragma unroll
                for (int i = 0; i < 8; ++i) vals[i] = rp[i];
            }
            scatter_bfrag(Hs, j, eb, vals);
        }
        __syncthreads();
        v16bf ac = load_a_frag40(Cs, w * 16);
        v16bf as = load_a_frag40(Ss, w * 16);
        v16bf h0 = load_frag32B(Hs + (((0 << 5) + lane) << 4));
        v16bf h1 = load_frag32B(Hs + (((1 << 5) + lane) << 4));
        v16bf h2 = load_frag32B(Hs + (((2 << 5) + lane) << 4));
        v16bf h3 = load_frag32B(Hs + (((3 << 5) + lane) << 4));
        accR[0] = wmma_bf16(ac, h0, accR[0]);
        accR[1] = wmma_bf16(ac, h1, accR[1]);
        accR[2] = wmma_bf16(ac, h2, accR[2]);
        accR[3] = wmma_bf16(ac, h3, accR[3]);
        accI[0] = wmma_bf16(as, h0, accI[0]);
        accI[1] = wmma_bf16(as, h1, accI[1]);
        accI[2] = wmma_bf16(as, h2, accI[2]);
        accI[3] = wmma_bf16(as, h3, accI[3]);
    }

    size_t base = (size_t)b * K * E;
#pragma unroll
    for (int nt = 0; nt < 4; ++nt)
#pragma unroll
        for (int r = 0; r < 8; ++r) {
            int kr = w * 16 + ((lane < 16) ? r : r + 8);
            int e  = e0 + nt * 16 + (lane & 15);
            float kv = kf[kr * E + e];
            size_t o = base + frag_off(kr, e, E);   // fragment order for hupdate
            Fr[o] = bft(accR[nt][r] * kv);
            Fi[o] = bft(accI[nt][r] * kv);
        }
}

// ---------------------------------------------------------------------------
// h_update[n,e] = sum_k c[n,k]*Fr[b,k,e] + s[n,k]*Fi[b,k,e]  (0.01*kf in F)
// Fr/Fi are fragment-ordered -> B fragments come straight from global.
// grid = (E/64, B, maxChunks of 128 atoms); rows past segment end are zero.
// ---------------------------------------------------------------------------
__global__ __launch_bounds__(256) void hupdate_kernel(
    const unsigned short* __restrict__ cM, const unsigned short* __restrict__ sM,
    const unsigned short* __restrict__ Fr, const unsigned short* __restrict__ Fi,
    const int* __restrict__ seg,
    unsigned short* __restrict__ hup, int N, int K, int E) {
    __shared__ __align__(16) unsigned short Cs[128 * 40];
    __shared__ __align__(16) unsigned short Ss[128 * 40];
    int b  = blockIdx.y;
    int s0 = seg[b], s1 = seg[b + 1];
    int r0 = s0 + blockIdx.z * 128;
    if (r0 >= s1) return;                 // uniform per block
    int e0 = blockIdx.x * 64;
    int t = threadIdx.x, w = t >> 5, lane = t & 31;
    size_t fbase = (size_t)b * K * E;
    int ntiles = E >> 4;
    v8f acc[4] = {};

    for (int kt = 0; kt < K; kt += 32) {
        __syncthreads();
        {   // A tiles (c,s): 128(atoms) x 32(k); row-granular masking, aligned
            int rr = t >> 1, jb = (t & 1) * 16;
            int n = r0 + rr;
            FragU fc, fs;
            if (n < s1) {
                const u32x4* pc = (const u32x4*)(cM + (size_t)n * K + kt + jb);
                const u32x4* ps = (const u32x4*)(sM + (size_t)n * K + kt + jb);
                fc.q[0] = pc[0]; fc.q[1] = pc[1];
                fs.q[0] = ps[0]; fs.q[1] = ps[1];
            } else {
#pragma unroll
                for (int i = 0; i < 8; ++i) { fc.w[i] = 0u; fs.w[i] = 0u; }
            }
            u32x4* dc = (u32x4*)(Cs + rr * 40 + jb);
            u32x4* ds = (u32x4*)(Ss + rr * 40 + jb);
            dc[0] = fc.q[0]; dc[1] = fc.q[1];
            ds[0] = fs.q[0]; ds[1] = fs.q[1];
        }
        __syncthreads();
        size_t bo = fbase + ((size_t)((kt >> 5) * ntiles + (e0 >> 4)) << 9)
                  + (lane << 4);
        v16bf r0f = load_frag32B(Fr + bo + (0 << 9));
        v16bf r1f = load_frag32B(Fr + bo + (1 << 9));
        v16bf r2f = load_frag32B(Fr + bo + (2 << 9));
        v16bf r3f = load_frag32B(Fr + bo + (3 << 9));
        v16bf i0f = load_frag32B(Fi + bo + (0 << 9));
        v16bf i1f = load_frag32B(Fi + bo + (1 << 9));
        v16bf i2f = load_frag32B(Fi + bo + (2 << 9));
        v16bf i3f = load_frag32B(Fi + bo + (3 << 9));
        v16bf af = load_a_frag40(Cs, w * 16);
        v16bf ag = load_a_frag40(Ss, w * 16);
        acc[0] = wmma_bf16(af, r0f, acc[0]);
        acc[1] = wmma_bf16(af, r1f, acc[1]);
        acc[2] = wmma_bf16(af, r2f, acc[2]);
        acc[3] = wmma_bf16(af, r3f, acc[3]);
        acc[0] = wmma_bf16(ag, i0f, acc[0]);
        acc[1] = wmma_bf16(ag, i1f, acc[1]);
        acc[2] = wmma_bf16(ag, i2f, acc[2]);
        acc[3] = wmma_bf16(ag, i3f, acc[3]);
    }

#pragma unroll
    for (int nt = 0; nt < 4; ++nt)
#pragma unroll
        for (int r = 0; r < 8; ++r) {
            int n = r0 + w * 16 + ((lane < 16) ? r : r + 8);
            if (n < s1)
                hup[(size_t)n * E + e0 + nt * 16 + (lane & 15)] = bft(acc[nt][r]);
        }
}

// ---------------------------------------------------------------------------
// Host-side orchestration
// ---------------------------------------------------------------------------
extern "C" void kernel_launch(void* const* d_in, const int* in_sizes, int n_in,
                              void* d_out, int out_size, void* d_ws, size_t ws_size,
                              hipStream_t stream) {
    (void)in_sizes; (void)n_in; (void)out_size; (void)ws_size;
    const int N = 4096, B = 64, KV = 128, E = 256, D = 8;

    const float* h     = (const float*)d_in[0];
    const float* x     = (const float*)d_in[1];
    const float* kvec  = (const float*)d_in[2];
    const int*   bseg  = (const int*)d_in[3];
    const float* Wpre  = (const float*)d_in[5];
    const float* Wdown = (const float*)d_in[6];
    const float* Wup   = (const float*)d_in[7];
    const float* W0    = (const float*)d_in[8];
    const float* Wres  = (const float*)d_in[9];

    char* ws = (char*)d_ws;
    size_t off = 0;
    auto alloc = [&](size_t bytes) -> void* {
        void* p = ws + off;
        off = (off + bytes + 255) & ~(size_t)255;
        return p;
    };
    unsigned short* wt   = (unsigned short*)alloc((size_t)9 * E * E * 2);
    float*          kf   = (float*)alloc((size_t)KV * E * 4);
    int*            seg  = (int*)alloc((size_t)(B + 1) * 4);
    unsigned short* hb   = (unsigned short*)alloc((size_t)N * E * 2);
    unsigned short* cM   = (unsigned short*)alloc((size_t)N * KV * 2);
    unsigned short* sM   = (unsigned short*)alloc((size_t)N * KV * 2);
    unsigned short* cT   = (unsigned short*)alloc((size_t)KV * N * 2);
    unsigned short* sT   = (unsigned short*)alloc((size_t)KV * N * 2);
    unsigned short* hres = (unsigned short*)alloc((size_t)N * E * 2);
    unsigned short* Fr   = (unsigned short*)alloc((size_t)B * KV * E * 2);
    unsigned short* Fi   = (unsigned short*)alloc((size_t)B * KV * E * 2);
    unsigned short* hup  = (unsigned short*)alloc((size_t)N * E * 2);
    unsigned short* t0   = (unsigned short*)alloc((size_t)N * E * 2);
    unsigned short* t1   = (unsigned short*)alloc((size_t)N * E * 2);
    unsigned short* t2   = (unsigned short*)alloc((size_t)N * E * 2);

    float* out_hu  = (float*)d_out;                 // (N,E)
    float* out_dot = (float*)d_out + (size_t)N * E; // (N,K)

    // --- prep ---
    seg_offsets_kernel<<<(N + 255) / 256, 256, 0, stream>>>(bseg, seg, N, B);
    f32_to_bf16_kernel<<<(N * E / 8 + 255) / 256, 256, 0, stream>>>(h, hb, N * E / 8);
    const float* wsrc[9] = {
        Wpre, Wpre + (size_t)E * E, W0,
        Wres + 0 * (size_t)E * E, Wres + 1 * (size_t)E * E,
        Wres + 2 * (size_t)E * E, Wres + 3 * (size_t)E * E,
        Wres + 4 * (size_t)E * E, Wres + 5 * (size_t)E * E };
    for (int i = 0; i < 9; ++i)
        wtrans_kernel<<<(E * E) / 256, 256, 0, stream>>>(wsrc[i], wt + (size_t)i * E * E, E);
    kfilter_kernel<<<(KV * E + 255) / 256, 256, 0, stream>>>(Wup, Wdown, kf, KV, E, D);
    dot_cs_kernel<<<(N * KV + 255) / 256, 256, 0, stream>>>(x, kvec, bseg, out_dot,
                                                            cM, sM, cT, sT, N, KV);

    dim3 gg(E / 64, N / 128);
    // hres = h + silu( silu(h @ Wpre0^T) @ Wpre1^T )
    gemm_silu_kernel<<<gg, 256, 0, stream>>>(hb, wt + 0 * (size_t)E * E,
                                             nullptr, nullptr, nullptr, t0, N, E, E);
    gemm_silu_kernel<<<gg, 256, 0, stream>>>(t0, wt + 1 * (size_t)E * E,
                                             h, nullptr, nullptr, hres, N, E, E);

    // structure factors + filter, then re-projection to atoms
    sf_kernel<<<dim3(E / 64, B), 256, 0, stream>>>(cT, sT, hres, kf, seg, Fr, Fi, N, KV, E);
    hupdate_kernel<<<dim3(E / 64, B, N / 128), 256, 0, stream>>>(cM, sM, Fr, Fi, seg,
                                                                 hup, N, KV, E);

    // hu = silu(h_update @ W0^T), then 3 residual blocks
    gemm_silu_kernel<<<gg, 256, 0, stream>>>(hup, wt + 2 * (size_t)E * E,
                                             nullptr, nullptr, nullptr, t0, N, E, E);
    unsigned short* xcur = t0;
    unsigned short* fa = t1;
    unsigned short* fb = t2;
    for (int i = 0; i < 3; ++i) {
        unsigned short* wr0 = wt + (size_t)(3 + 2 * i) * E * E;
        unsigned short* wr1 = wt + (size_t)(4 + 2 * i) * E * E;
        gemm_silu_kernel<<<gg, 256, 0, stream>>>(xcur, wr0,
                                                 nullptr, nullptr, nullptr, fa, N, E, E);
        if (i < 2) {
            gemm_silu_kernel<<<gg, 256, 0, stream>>>(fa, wr1,
                                                     nullptr, xcur, nullptr, fb, N, E, E);
            unsigned short* old = xcur; xcur = fb; fb = fa; fa = old;
        } else {
            gemm_silu_kernel<<<gg, 256, 0, stream>>>(fa, wr1,
                                                     nullptr, xcur, out_hu, nullptr, N, E, E);
        }
    }
}
